// sparseKT_35450660061905
// MI455X (gfx1250) — matrix operand order repaired and checked by reading, not compiled, not run on hardware
//
#include <hip/hip_runtime.h>

// ============================================================================
// Sparse-top-k multi-head attention forward for MI455X (gfx1250, wave32, WMMA)
//   BS=32, S=512, D=512, H=8, DK=64, k_index from device scalar
// GEMMs: v_wmma_f32_16x16x32_f16 (f16 A/B, f32 accumulate), operands
// pre-converted to f16 once. Attention V strip staged into LDS with the
// Tensor Data Mover (tensor_load_to_lds), overlapped with QK^T + softmax,
// synchronized with s_wait_tensorcnt.
// ============================================================================

typedef __attribute__((ext_vector_type(16))) _Float16 v16h;
typedef __attribute__((ext_vector_type(8)))  float    v8f;
typedef unsigned int __attribute__((ext_vector_type(4))) u32x4;
typedef int __attribute__((ext_vector_type(4)))          i32x4;
typedef int __attribute__((ext_vector_type(8)))          i32x8;

#define BSZ    32
#define SEQ    512
#define DMODEL 512
#define NHEAD  8
#define DKH    64
#define NEGBIG (-1e32f)

#if defined(__has_builtin)
#if __has_builtin(__builtin_amdgcn_tensor_load_to_lds)
#define HAVE_TDM 1
#endif
#endif
// This toolchain (clang-23 / therock-10.0) uses the 6-arg builtin form
// (probe: "too few arguments to function call, expected 6, have 5").
#ifndef TDM_SIX_ARGS
#define TDM_SIX_ARGS 1
#endif

// 16-bit A-matrix 16x32 fragment K-index map (contiguous, matches B):
// lanes 0-15 (g=0): elements 0..15 -> K 0..15
// lanes 16-31(g=1): elements 0..15 -> K 16..31
__device__ __forceinline__ int kmapA(int g, int j) {
  return g * 16 + j;
}

__device__ __forceinline__ float waveMax(float v) {
#pragma unroll
  for (int o = 16; o; o >>= 1) v = fmaxf(v, __shfl_xor(v, o, 32));
  return v;
}
__device__ __forceinline__ float waveSum(float v) {
#pragma unroll
  for (int o = 16; o; o >>= 1) v += __shfl_xor(v, o, 32);
  return v;
}

// wave-parallel softmax over one LDS row of SEQ floats (lane owns c%32==lane)
__device__ __forceinline__ void softmax_row(float* row, int lane) {
  float m = -3.4e38f;
  for (int c = lane; c < SEQ; c += 32) m = fmaxf(m, row[c]);
  m = waveMax(m);
  float s = 0.f;
  for (int c = lane; c < SEQ; c += 32) {
    float e = __expf(row[c] - m);
    row[c] = e;
    s += e;
  }
  s = waveSum(s);
  float inv = 1.f / s;
  for (int c = lane; c < SEQ; c += 32) row[c] *= inv;
}

// ---------------------------------------------------------------------------
// Bulk f32 -> f16 conversion (one pass; n multiple of 4)
// ---------------------------------------------------------------------------
__global__ void cvt_f32_f16_kernel(const float* __restrict__ x,
                                   _Float16* __restrict__ y, long n) {
  const long i0 = ((long)blockIdx.x * blockDim.x + threadIdx.x) * 4;
  const long stride = (long)gridDim.x * blockDim.x * 4;
  for (long i = i0; i < n; i += stride) {
    float4 f = *(const float4*)(x + i);
    y[i + 0] = (_Float16)f.x;
    y[i + 1] = (_Float16)f.y;
    y[i + 2] = (_Float16)f.z;
    y[i + 3] = (_Float16)f.w;
  }
}

// ---------------------------------------------------------------------------
// Y(headsplit f16) = Xh(f16) * Wh^T(f16) + b(f32)
// Xh: (BS*SEQ, D) row-major; Wh: (D_out, D_in) row-major; Y: (BS, H, SEQ, DK)
// One wave computes a 16x64 output tile.
// ---------------------------------------------------------------------------
__global__ void proj_kernel(const _Float16* __restrict__ Xh,
                            const _Float16* __restrict__ Wh,
                            const float* __restrict__ bias,
                            _Float16* __restrict__ Y) {
  const int wave = (int)((blockIdx.x * blockDim.x + threadIdx.x) >> 5);
  const int lane = (int)(threadIdx.x & 31);
  const int rowTile  = wave >> 3;   // 1024 tiles of 16 rows
  const int colGroup = wave & 7;    // 8 groups of 64 cols
  const int g = lane >> 4;
  const int r = lane & 15;

  v8f acc[4];
#pragma unroll
  for (int ct = 0; ct < 4; ++ct)
#pragma unroll
    for (int i = 0; i < 8; ++i) acc[ct][i] = 0.f;

  const long rowBase = (long)rowTile * 16;
  const int  nBase   = colGroup * 64;
  const _Float16* arow = Xh + (rowBase + r) * DMODEL;

  for (int kk = 0; kk < DMODEL; kk += 32) {
    __builtin_prefetch(arow + kk + 64, 0, 0);   // global_prefetch_b8
    v16h a;
#pragma unroll
    for (int j = 0; j < 16; ++j) a[j] = arow[kk + kmapA(g, j)];
#pragma unroll
    for (int ct = 0; ct < 4; ++ct) {
      const int n = nBase + ct * 16 + r;        // B column owned by this lane
      v16h bf;
#pragma unroll
      for (int j = 0; j < 16; ++j)              // B[k][n] = W[n][k], contiguous
        bf[j] = Wh[(long)n * DMODEL + kk + g * 16 + j];
      acc[ct] = __builtin_amdgcn_wmma_f32_16x16x32_f16(
          false, a, false, bf, (short)0, acc[ct], false, false);
    }
  }

#pragma unroll
  for (int ct = 0; ct < 4; ++ct) {
    const int n  = nBase + ct * 16 + r;
    const int h  = n >> 6;
    const int dk = n & 63;
    const float bv = bias[n];
#pragma unroll
    for (int i = 0; i < 8; ++i) {
      const long m  = rowBase + i + g * 8;
      const long b_ = m >> 9;       // / SEQ
      const long s_ = m & 511;
      Y[(((b_ * NHEAD) + h) * SEQ + s_) * DKH + dk] = (_Float16)(acc[ct][i] + bv);
    }
  }
}

// ---------------------------------------------------------------------------
// Attention: one wave per (batch, head, 16-row block).
// TDM prefetch of V strip into LDS overlapped with:
// scores -> softmax -> top-k sparsify -> re-softmax -> zero row0; then PV.
// ---------------------------------------------------------------------------
__global__ __launch_bounds__(32) void attn_kernel(const _Float16* __restrict__ Qh,
                                                  const _Float16* __restrict__ Kh,
                                                  const _Float16* __restrict__ Vh,
                                                  const int* __restrict__ kidx_p,
                                                  _Float16* __restrict__ CC) {
  __shared__ float sc[16][SEQ];            // 32 KB score strip
  __shared__ float tmp[SEQ];               // top-k scratch row
  __shared__ _Float16 Vt[SEQ * DKH];       // 64 KB V strip (TDM destination)

  const int lane = (int)threadIdx.x;
  const int g = lane >> 4;
  const int r = lane & 15;
  const int blk = (int)blockIdx.x;
  const int rowTile = blk & 31;
  const int h = (blk >> 5) & 7;
  const int b = blk >> 8;
  const long base = ((long)b * NHEAD + h) * SEQ * DKH;
  const int kidx = *kidx_p;
  const int rowBase = rowTile * 16;

#if HAVE_TDM
  {
    // 2-D tensor descriptor: 64 x 512 strip of 2-byte elements, row stride 64.
    const unsigned long long gaddr = (unsigned long long)(const void*)(Vh + base);
    const unsigned lds_off = (unsigned)(uintptr_t)(&Vt[0]);
    u32x4 g0;
    g0[0] = 1u;                                         // count=1, user D#
    g0[1] = lds_off;                                    // lds_addr
    g0[2] = (unsigned)(gaddr & 0xffffffffu);            // global_addr[31:0]
    g0[3] = (unsigned)((gaddr >> 32) & 0x01ffffffu)     // global_addr[56:32]
            | (2u << 30);                               // type = 2 (image)
    i32x8 g1;
    g1[0] = (1 << 16);            // workgroup_mask=0, data_size=1 (2 bytes)
    g1[1] = (int)(DKH << 16);     // tensor_dim0[15:0] = 64
    g1[2] = (int)(SEQ << 16);     // tensor_dim0 hi=0 | tensor_dim1 lo = 512
    g1[3] = (int)(DKH << 16);     // tensor_dim1 hi=0 | tile_dim0 = 64
    g1[4] = SEQ;                  // tile_dim1 = 512, tile_dim2 = 0
    g1[5] = DKH;                  // tensor_dim0_stride = 64
    g1[6] = 0;                    // stride hi | tensor_dim1_stride lo
    g1[7] = 0;
    i32x4 z4;
    z4[0] = z4[1] = z4[2] = z4[3] = 0;
#if TDM_SIX_ARGS
    i32x8 z8;
#pragma unroll
    for (int i = 0; i < 8; ++i) z8[i] = 0;
    __builtin_amdgcn_tensor_load_to_lds(g0, g1, z4, z4, z8, 0);
#else
    __builtin_amdgcn_tensor_load_to_lds(g0, g1, z4, z4, 0);
#endif
  }
#else
  // Fallback: cooperative 16-byte copies of the V strip into LDS.
  for (int i = lane; i < (SEQ * DKH) / 8; i += 32)
    ((float4*)Vt)[i] = ((const float4*)(Vh + base))[i];
#endif

  // Q fragments: 16 rows x 64 (two 32-wide K chunks)
  v16h a0, a1;
#pragma unroll
  for (int j = 0; j < 16; ++j) {
    const long qoff = base + (long)(rowBase + r) * DKH;
    a0[j] = Qh[qoff + kmapA(g, j)];
    a1[j] = Qh[qoff + 32 + kmapA(g, j)];
  }

  // ---- scores = Q K^T / 8, causal mask ----
  for (int jt = 0; jt < SEQ / 16; ++jt) {
    if (jt > rowTile) {  // fully masked tile (uniform branch)
#pragma unroll
      for (int i = 0; i < 8; ++i) sc[i + g * 8][jt * 16 + r] = NEGBIG;
      continue;
    }
    v16h b0, b1;
#pragma unroll
    for (int j = 0; j < 16; ++j) {   // B[d][n] = Kh[jt*16+n][d]
      const long koff = base + (long)(jt * 16 + r) * DKH;
      b0[j] = Kh[koff + g * 16 + j];
      b1[j] = Kh[koff + 32 + g * 16 + j];
    }
    v8f c;
#pragma unroll
    for (int i = 0; i < 8; ++i) c[i] = 0.f;
    c = __builtin_amdgcn_wmma_f32_16x16x32_f16(false, a0, false, b0, (short)0, c, false, false);
    c = __builtin_amdgcn_wmma_f32_16x16x32_f16(false, a1, false, b1, (short)0, c, false, false);
#pragma unroll
    for (int i = 0; i < 8; ++i) {
      const int M = i + g * 8;
      const int row = rowBase + M;
      const int col = jt * 16 + r;
      float v = c[i] * 0.125f;           // 1/sqrt(64)
      if (col > row) v = NEGBIG;
      sc[M][col] = v;
    }
  }
  __syncthreads();

  // ---- first softmax (all rows) ----
  for (int i = 0; i < 16; ++i) softmax_row(sc[i], lane);
  __syncthreads();

  // ---- sparsify rows with global row >= k_index ----
  for (int i = 0; i < 16; ++i) {
    const int grow = rowBase + i;
    if (grow < kidx) continue;          // uniform branch
    for (int c = lane; c < SEQ; c += 32) tmp[c] = sc[i][c];
    __syncthreads();
    float thr = 0.f;
    for (int t = 0; t < kidx; ++t) {    // iterative argmax-and-kill
      float best = -3.4e38f;
      int bidx = SEQ;
      for (int c = lane; c < SEQ; c += 32) {
        float v = tmp[c];
        if (v > best) { best = v; bidx = c; }
      }
#pragma unroll
      for (int o = 16; o; o >>= 1) {
        float ov = __shfl_xor(best, o, 32);
        int   oi = __shfl_xor(bidx, o, 32);
        if (ov > best || (ov == best && oi < bidx)) { best = ov; bidx = oi; }
      }
      thr = best;
      if (lane == 0 && bidx < SEQ) tmp[bidx] = -3.4e38f;
      __syncthreads();
    }
    for (int c = lane; c < SEQ; c += 32) {
      float v = sc[i][c];
      sc[i][c] = (v - thr >= 0.f) ? v : NEGBIG;
    }
    __syncthreads();
  }

  // ---- second softmax (ALL rows, matching the reference quirk) ----
  for (int i = 0; i < 16; ++i) softmax_row(sc[i], lane);
  __syncthreads();

  // ---- zero_pad: row 0 of attention -> zeros ----
  if (rowTile == 0) {
    for (int c = lane; c < SEQ; c += 32) sc[0][c] = 0.f;
  }
  __syncthreads();

#if HAVE_TDM
  __builtin_amdgcn_s_wait_tensorcnt(0);   // V strip now resident in LDS
#endif
  __syncthreads();

  // ---- out = attn * V  (truncate K-loop at causal frontier) ----
  v8f o[4];
#pragma unroll
  for (int ct = 0; ct < 4; ++ct)
#pragma unroll
    for (int i = 0; i < 8; ++i) o[ct][i] = 0.f;

  const int kkTiles = (rowBase + 16 + 31) / 32;   // remaining attn cols are 0
  for (int kt = 0; kt < kkTiles; ++kt) {
    const int kk = kt * 32;
    v16h a;
#pragma unroll
    for (int j = 0; j < 16; ++j)
      a[j] = (_Float16)sc[r][kk + kmapA(g, j)];
#pragma unroll
    for (int ct = 0; ct < 4; ++ct) {
      v16h bf;
#pragma unroll
      for (int j = 0; j < 16; ++j)   // B[k][n] = Vt[kk+k][ct*16+n]
        bf[j] = Vt[(kk + g * 16 + j) * DKH + ct * 16 + r];
      o[ct] = __builtin_amdgcn_wmma_f32_16x16x32_f16(
          false, a, false, bf, (short)0, o[ct], false, false);
    }
  }

  // store concat: (b, s, h*64 + dk) as f16
#pragma unroll
  for (int ct = 0; ct < 4; ++ct) {
    const int dk = ct * 16 + r;
#pragma unroll
    for (int i = 0; i < 8; ++i) {
      const int s_ = rowBase + i + g * 8;
      CC[((long)b * SEQ + s_) * DMODEL + h * DKH + dk] = (_Float16)o[ct][i];
    }
  }
}

// ---------------------------------------------------------------------------
// Out(f32) = Concat(f16) * Woh^T(f16) + bo
// ---------------------------------------------------------------------------
__global__ void outproj_kernel(const _Float16* __restrict__ CC,
                               const _Float16* __restrict__ Wh,
                               const float* __restrict__ bias,
                               float* __restrict__ Out) {
  const int wave = (int)((blockIdx.x * blockDim.x + threadIdx.x) >> 5);
  const int lane = (int)(threadIdx.x & 31);
  const int rowTile  = wave >> 3;
  const int colGroup = wave & 7;
  const int g = lane >> 4;
  const int r = lane & 15;

  v8f acc[4];
#pragma unroll
  for (int ct = 0; ct < 4; ++ct)
#pragma unroll
    for (int i = 0; i < 8; ++i) acc[ct][i] = 0.f;

  const long rowBase = (long)rowTile * 16;
  const int  nBase   = colGroup * 64;
  const _Float16* arow = CC + (rowBase + r) * DMODEL;

  for (int kk = 0; kk < DMODEL; kk += 32) {
    __builtin_prefetch(arow + kk + 64, 0, 0);
    v16h a;
#pragma unroll
    for (int j = 0; j < 16; ++j) a[j] = arow[kk + kmapA(g, j)];
#pragma unroll
    for (int ct = 0; ct < 4; ++ct) {
      const int n = nBase + ct * 16 + r;
      v16h bf;
#pragma unroll
      for (int j = 0; j < 16; ++j)
        bf[j] = Wh[(long)n * DMODEL + kk + g * 16 + j];
      acc[ct] = __builtin_amdgcn_wmma_f32_16x16x32_f16(
          false, a, false, bf, (short)0, acc[ct], false, false);
    }
  }

#pragma unroll
  for (int ct = 0; ct < 4; ++ct) {
    const int n = nBase + ct * 16 + r;
    const float bv = bias[n];
#pragma unroll
    for (int i = 0; i < 8; ++i) {
      const long m = rowBase + i + g * 8;
      Out[m * DMODEL + n] = acc[ct][i] + bv;
    }
  }
}

// ---------------------------------------------------------------------------
extern "C" void kernel_launch(void* const* d_in, const int* in_sizes, int n_in,
                              void* d_out, int out_size, void* d_ws, size_t ws_size,
                              hipStream_t stream) {
  const float* q  = (const float*)d_in[0];
  const float* k  = (const float*)d_in[1];
  const float* v  = (const float*)d_in[2];
  const float* Wq = (const float*)d_in[3];
  const float* bq = (const float*)d_in[4];
  const float* Wk = (const float*)d_in[5];
  const float* bk = (const float*)d_in[6];
  const float* Wv = (const float*)d_in[7];
  const float* bv = (const float*)d_in[8];
  const float* Wo = (const float*)d_in[9];
  const float* bo = (const float*)d_in[10];
  const int* kidx = (const int*)d_in[11];

  const size_t NEL = (size_t)BSZ * SEQ * DMODEL;  // 8,388,608 elems
  const size_t WEL = (size_t)DMODEL * DMODEL;     // 262,144 elems
  _Float16* qh  = (_Float16*)d_ws;
  _Float16* kh  = qh + NEL;
  _Float16* vh  = kh + NEL;
  _Float16* cc  = vh + NEL;
  _Float16* xq  = cc + NEL;
  _Float16* xk  = xq + NEL;
  _Float16* xv  = xk + NEL;
  _Float16* wqh = xv + NEL;
  _Float16* wkh = wqh + WEL;
  _Float16* wvh = wkh + WEL;
  _Float16* woh = wvh + WEL;

  // one-time f32 -> f16 conversions
  const dim3 cblkX((unsigned)(NEL / 4 / 256)), cblkW((unsigned)(WEL / 4 / 256));
  cvt_f32_f16_kernel<<<cblkX, 256, 0, stream>>>(q, xq, (long)NEL);
  cvt_f32_f16_kernel<<<cblkX, 256, 0, stream>>>(k, xk, (long)NEL);
  cvt_f32_f16_kernel<<<cblkX, 256, 0, stream>>>(v, xv, (long)NEL);
  cvt_f32_f16_kernel<<<cblkW, 256, 0, stream>>>(Wq, wqh, (long)WEL);
  cvt_f32_f16_kernel<<<cblkW, 256, 0, stream>>>(Wk, wkh, (long)WEL);
  cvt_f32_f16_kernel<<<cblkW, 256, 0, stream>>>(Wv, wvh, (long)WEL);
  cvt_f32_f16_kernel<<<cblkW, 256, 0, stream>>>(Wo, woh, (long)WEL);

  // 8192 waves of 16x64 tiles; 4 waves (128 thr) per workgroup
  const dim3 projGrid(2048), projBlk(128);
  proj_kernel<<<projGrid, projBlk, 0, stream>>>(xq, wqh, bq, qh);
  proj_kernel<<<projGrid, projBlk, 0, stream>>>(xk, wkh, bk, kh);
  proj_kernel<<<projGrid, projBlk, 0, stream>>>(xv, wvh, bv, vh);

  // one wave per (batch, head, 16-row block): 32*8*32 = 8192 blocks
  attn_kernel<<<dim3(BSZ * NHEAD * (SEQ / 16)), dim3(32), 0, stream>>>(
      qh, kh, vh, kidx, cc);

  outproj_kernel<<<projGrid, projBlk, 0, stream>>>(cc, woh, bo, (float*)d_out);
}
